// StateQuantizer_38319698215679
// MI455X (gfx1250) — matrix operand order repaired and checked by reading, not compiled, not run on hardware
//
#include <hip/hip_runtime.h>

typedef __attribute__((ext_vector_type(16))) _Float16 v16h;
typedef __attribute__((ext_vector_type(8)))  _Float16 v8h;
typedef __attribute__((ext_vector_type(4)))  _Float16 v4h;
typedef __attribute__((ext_vector_type(8)))  float    v8f;
typedef __attribute__((ext_vector_type(4)))  float    v4f;

#define D_DIM 128
#define K_DIM 1024
#define BN_TOT 65536
#define ROWS_PER_BLOCK 128
#define THREADS 256
#define WAVES_PER_BLOCK (THREADS / 32)
#define NUM_BLOCKS (BN_TOT / ROWS_PER_BLOCK)      // 512

// Codebook image in LDS: 272-byte row stride (136 halves) -> lane-to-bank
// stride of 68 dwords == 4 (mod 64), so the 16 lanes of a half-group cover
// all 64 banks (conflict-free), instead of the 16-way conflict at stride 256B.
#define EH_STRIDE 136                              // halves per row
#define EH_ROWS   (K_DIM + 16)                     // +1 tile pad for prefetch overrun
#define ENORM_OFF (EH_ROWS * EH_STRIDE * 2)        // bytes
#define LDS_BYTES (ENORM_OFF + (K_DIM + 16) * 4)   // ~287 KB < 320 KB/WGP

__device__ __forceinline__ v16h load_bfrag(const _Float16* eh, int row, int c, int g)
{
    // B 32x16 f16 layout: lane = column; half g holds K {8g..8g+7, 16+8g..16+8g+7}.
    const _Float16* p = eh + row * EH_STRIDE + c * 32 + g * 8;
    v8h lo = *(const v8h*)p;          // 16B aligned (stride 272 = 17*16)
    v8h hi = *(const v8h*)(p + 16);
    return __builtin_shufflevector(lo, hi, 0,1,2,3,4,5,6,7,8,9,10,11,12,13,14,15);
}

__global__ void __launch_bounds__(THREADS)
vq_main_kernel(const float* __restrict__ z, const float* __restrict__ E,
               float* __restrict__ out, float* __restrict__ partials)
{
    extern __shared__ __align__(32) char smem[];
    _Float16* eh    = (_Float16*)smem;
    float*    enorm = (float*)(smem + ENORM_OFF);

    const int tid  = threadIdx.x;
    const int lane = tid & 31;
    const int wave = tid >> 5;
    const int g    = lane >> 4;
    const int l15  = lane & 15;

    // ---- Stage 1: codebook -> LDS f16 (padded stride) + ||e_k||^2
    for (int r = 0; r < K_DIM / WAVES_PER_BLOCK; ++r) {
        const int row = wave * (K_DIM / WAVES_PER_BLOCK) + r;
        v4f v = *(const v4f*)(E + row * D_DIM + lane * 4);
        float ss = v[0]*v[0] + v[1]*v[1] + v[2]*v[2] + v[3]*v[3];
        #pragma unroll
        for (int off = 16; off > 0; off >>= 1) ss += __shfl_xor(ss, off, 32);
        if (lane == 0) enorm[row] = ss;
        v4h h; h[0] = (_Float16)v[0]; h[1] = (_Float16)v[1];
               h[2] = (_Float16)v[2]; h[3] = (_Float16)v[3];
        *(v4h*)(eh + row * EH_STRIDE + lane * 4) = h;
    }
    __syncthreads();

    // ---- Stage 2: per-wave A fragments (16 rows x 128 d) in registers
    // A 16x32 f16 layout: lane = row; lanes<16 hold K {c*32+0..7, +16..23},
    // lanes>=16 hold K {c*32+8..15, +24..31}.
    const int waveRowBase = blockIdx.x * ROWS_PER_BLOCK + wave * 16;
    const int arow = waveRowBase + l15;
    v16h afrag[4];
    #pragma unroll
    for (int c = 0; c < 4; ++c) {
        const int b1 = c * 32 + g * 8;
        const int b2 = c * 32 + 16 + g * 8;
        v4f f0 = *(const v4f*)(z + arow * D_DIM + b1);
        v4f f1 = *(const v4f*)(z + arow * D_DIM + b1 + 4);
        v4f f2 = *(const v4f*)(z + arow * D_DIM + b2);
        v4f f3 = *(const v4f*)(z + arow * D_DIM + b2 + 4);
        #pragma unroll
        for (int i = 0; i < 4; ++i) {
            afrag[c][i]      = (_Float16)f0[i];
            afrag[c][4 + i]  = (_Float16)f1[i];
            afrag[c][8 + i]  = (_Float16)f2[i];
            afrag[c][12 + i] = (_Float16)f3[i];
        }
    }

    float best[8];
    int   bidx[8];
    #pragma unroll
    for (int j = 0; j < 8; ++j) { best[j] = 3.4e38f; bidx[j] = 0; }

    // ---- Stage 3: 64 K-tiles, double-buffered B fragments + norms.
    // score_k = ||e_k||^2 - 2 * (z . e_k)   (||z||^2 dropped: constant per row)
    v16h b0[4], b1[4];
    #pragma unroll
    for (int c = 0; c < 4; ++c) b0[c] = load_bfrag(eh, l15, c, g);
    float en0 = enorm[l15];
    float en1;

    for (int t = 0; t < K_DIM / 16; t += 2) {
        // prefetch tile t+1 while XDL computes tile t
        #pragma unroll
        for (int c = 0; c < 4; ++c) b1[c] = load_bfrag(eh, (t + 1) * 16 + l15, c, g);
        en1 = enorm[(t + 1) * 16 + l15];
        {
            v8f acc = {};
            #pragma unroll
            for (int c = 0; c < 4; ++c)
                acc = __builtin_amdgcn_wmma_f32_16x16x32_f16(
                    false, afrag[c], false, b0[c], (short)0, acc, false, false);
            #pragma unroll
            for (int j = 0; j < 8; ++j) {
                const float score = fmaf(-2.0f, acc[j], en0);
                if (score < best[j]) { best[j] = score; bidx[j] = t * 16 + l15; }
            }
        }
        // prefetch tile t+2 (last iteration reads the padded rows; unused)
        #pragma unroll
        for (int c = 0; c < 4; ++c) b0[c] = load_bfrag(eh, (t + 2) * 16 + l15, c, g);
        en0 = enorm[(t + 2) * 16 + l15];
        {
            v8f acc = {};
            #pragma unroll
            for (int c = 0; c < 4; ++c)
                acc = __builtin_amdgcn_wmma_f32_16x16x32_f16(
                    false, afrag[c], false, b1[c], (short)0, acc, false, false);
            #pragma unroll
            for (int j = 0; j < 8; ++j) {
                const float score = fmaf(-2.0f, acc[j], en1);
                if (score < best[j]) { best[j] = score; bidx[j] = (t + 1) * 16 + l15; }
            }
        }
    }

    // ---- Stage 4: argmin across the 16 lanes of each C half-group
    #pragma unroll
    for (int j = 0; j < 8; ++j) {
        #pragma unroll
        for (int off = 1; off < 16; off <<= 1) {
            const float ob = __shfl_xor(best[j], off, 32);
            const int   oi = __shfl_xor(bidx[j], off, 32);
            if (ob < best[j] || (ob == best[j] && oi < bidx[j])) {
                best[j] = ob; bidx[j] = oi;
            }
        }
    }

    // ---- Stage 5: gather exact f32 codebook rows -> out; loss partial.
    // C layout: lanes 0-15 hold rows M=j, lanes 16-31 rows M=8+j.
    float lacc = 0.0f;
    #pragma unroll
    for (int j = 0; j < 8; ++j) {
        const int gr  = waveRowBase + g * 8 + j;
        const int idx = bidx[j];                 // uniform within 16-lane group
        const int s   = l15 * 8;                 // 16 lanes x 8 floats = 128
        v4f e0 = *(const v4f*)(E + idx * D_DIM + s);
        v4f e1 = *(const v4f*)(E + idx * D_DIM + s + 4);
        v4f z0 = *(const v4f*)(z + gr * D_DIM + s);
        v4f z1 = *(const v4f*)(z + gr * D_DIM + s + 4);
        *(v4f*)(out + gr * D_DIM + s)     = e0;
        *(v4f*)(out + gr * D_DIM + s + 4) = e1;
        #pragma unroll
        for (int i = 0; i < 4; ++i) {
            const float d0 = e0[i] - z0[i];
            const float d1 = e1[i] - z1[i];
            lacc = fmaf(d0, d0, lacc);
            lacc = fmaf(d1, d1, lacc);
        }
    }
    #pragma unroll
    for (int off = 16; off > 0; off >>= 1) lacc += __shfl_xor(lacc, off, 32);
    if (lane == 0) partials[blockIdx.x * WAVES_PER_BLOCK + wave] = lacc;
}

// Deterministic fixed-order reduction of the 4096 wave partials -> scalar loss.
__global__ void vq_loss_reduce_kernel(const float* __restrict__ partials,
                                      float* __restrict__ out_loss)
{
    if (threadIdx.x == 0 && blockIdx.x == 0) {
        float s = 0.0f;
        for (int i = 0; i < NUM_BLOCKS * WAVES_PER_BLOCK; ++i) s += partials[i];
        // KLD_SCALE*(COMMITMENT_COST*mse + mse) = 10*1.25*mse
        *out_loss = 12.5f * (s / (float)(BN_TOT * D_DIM));
    }
}

extern "C" void kernel_launch(void* const* d_in, const int* in_sizes, int n_in,
                              void* d_out, int out_size, void* d_ws, size_t ws_size,
                              hipStream_t stream)
{
    (void)in_sizes; (void)n_in; (void)out_size; (void)ws_size;
    const float* z = (const float*)d_in[0];   // (32, 2048, 128) f32
    const float* E = (const float*)d_in[1];   // (1024, 128) f32
    float* out      = (float*)d_out;          // 65536*128 z_q + 1 loss
    float* partials = (float*)d_ws;           // 4096 floats

    (void)hipFuncSetAttribute((const void*)vq_main_kernel,
                              hipFuncAttributeMaxDynamicSharedMemorySize,
                              LDS_BYTES);

    vq_main_kernel<<<NUM_BLOCKS, THREADS, LDS_BYTES, stream>>>(z, E, out, partials);
    vq_loss_reduce_kernel<<<1, 32, 0, stream>>>(partials, out + (size_t)BN_TOT * D_DIM);
}
